// ConvLSTM_34754875359605
// MI455X (gfx1250) — matrix operand order repaired and checked by reading, not compiled, not run on hardware
//
#include <hip/hip_runtime.h>

// ---------------------------------------------------------------------------
// CDNA5 (gfx1250, wave32) ConvLSTM stack + MLP head, bf16 WMMA everywhere.
// - Conv weights pre-packed per launch into bf16 (COUT, K32), K zero-padded to
//   a multiple of 32: each lane's WMMA B fragment is one contiguous 32-byte
//   global load (no LDS staging, no guards, no cvt in the hot loop).
// - All activations (x, h) are bf16; LSTM cell state c and GEMM accumulation
//   stay fp32.  A tiles are staged bf16->LDS as (m,k) rows; each lane reads
//   16 contiguous k, k = 16h + s (two ds_load_b128 per fragment), branch-free, conversion-free.
// ---------------------------------------------------------------------------

typedef __attribute__((ext_vector_type(16))) __bf16 bf16x16;
typedef __attribute__((ext_vector_type(8)))  __bf16 bf16x8;
typedef __attribute__((ext_vector_type(4)))  __bf16 bf16x4;
typedef __attribute__((ext_vector_type(2)))  __bf16 bf16x2;
typedef __attribute__((ext_vector_type(8)))  float  f32x8;

__device__ __forceinline__ float hard_sigmoid(float z) {
  return fminf(fmaxf(0.2f * z + 0.5f, 0.f), 1.f);
}
__device__ __forceinline__ int imin(int a, int b) { return a < b ? a : b; }
__device__ __forceinline__ int imax(int a, int b) { return a > b ? a : b; }

// Build a 16-wide A fragment from the two contiguous 8-element K runs.
__device__ __forceinline__ bf16x16 cat8(bf16x8 lo, bf16x8 hi) {
  return __builtin_shufflevector(lo, hi, 0, 1, 2, 3, 4, 5, 6, 7,
                                 8, 9, 10, 11, 12, 13, 14, 15);
}

// -------------------------------- utility ----------------------------------
__global__ void fill_zero_kernel(float* __restrict__ p, long n) {
  long i = (long)blockIdx.x * blockDim.x + threadIdx.x;
  if (i < n) p[i] = 0.f;
}

__global__ void f32_to_bf16_kernel(const float* __restrict__ src,
                                   __bf16* __restrict__ dst, long n) {
  long i = (long)blockIdx.x * blockDim.x + threadIdx.x;
  if (i < n) dst[i] = (__bf16)src[i];
}

// Pack fp32 (KTOT, COUT) weights -> bf16 (COUT, K32), zero-padded K rows.
__global__ void pack_weight_kernel(const float* __restrict__ wgt,
                                   __bf16* __restrict__ out,
                                   int KTOT, int K32, int COUT, long nelem) {
  long idx = (long)blockIdx.x * blockDim.x + threadIdx.x;
  if (idx >= nelem) return;
  const int n = (int)(idx % COUT);          // coalesced read side
  const int k = (int)(idx / COUT);
  float v = (k < KTOT) ? wgt[(long)k * COUT + n] : 0.f;
  out[(long)n * K32 + k] = (__bf16)v;
}

// ------------------------- implicit-GEMM 5x5 conv --------------------------
// out(m, n) = sum_k A(m,k) * W(k,n), m = img*Ho*Wo + oy*Wo + ox,
// k = (ky*5+kx)*CIN + ci, n = cout.  PAD=0 -> VALID, PAD=2 -> SAME.
// Block: 256 thr = 8 waves as 2(M) x 4(N); block tile M=32, N=128.
// Requires Mtot % 32 == 0 and COUT % 128 == 0 (true for all call sites).
template <int CIN, int COUT, int PAD>
__global__ __launch_bounds__(256)
void conv5x5_wmma(const __bf16* __restrict__ in, long in_img_stride,
                  const __bf16* __restrict__ wgtT,  // (COUT, K32) bf16 packed
                  const float* __restrict__ bias,   // COUT or nullptr
                  float* __restrict__ out,          // (Mtot, COUT)
                  int H, int W, int Ho, int Wo, int accum) {
  constexpr int  KTOT = 25 * CIN;
  constexpr int  KCH  = (KTOT + 31) / 32;
  constexpr int  K32  = KCH * 32;
  constexpr bool KG   = (KTOT % 32) != 0;   // only CIN=5 (K=125)

  __shared__ __bf16 As[32][40];  // (m, k), 8-bf16 row pad -> 80B row stride

  const int tid  = threadIdx.x;
  const int lane = tid & 31;
  const int wave = tid >> 5;
  const int wm   = wave >> 2;   // 0..1  (M sub-tile)
  const int wn   = wave & 3;    // 0..3  (N group: 2 sub-tiles each)

  const int HoWo = Ho * Wo;
  const int m0 = blockIdx.x * 32;
  const int n0 = blockIdx.y * 128;

  // A staging: 4 consecutive k of one row per thread.
  const int a_mi = tid >> 3;            // 0..31
  const int a_kk = (tid & 7) * 4;       // 0,4,...,28
  const int a_m  = m0 + a_mi;
  const int a_img = a_m / HoWo;
  const int a_p   = a_m - a_img * HoWo;
  const int a_oy  = a_p / Wo;
  const int a_ox  = a_p - a_oy * Wo;
  const __bf16* a_base = in + (long)a_img * in_img_stride;

  // B fragments read directly from packed global weights: per lane one
  // contiguous 32B run (n column, 16 consecutive k) per fragment.
  const int ncol = wn * 32 + (lane & 15);
  const int kbB  = (lane < 16) ? 0 : 16;
  const __bf16* bptr0 = wgtT + (long)(n0 + ncol) * K32 + kbB;
  const __bf16* bptr1 = bptr0 + (long)16 * K32;

  f32x8 acc0 = {};
  f32x8 acc1 = {};

  for (int kc = 0; kc < KCH; ++kc) {
    const int k0 = kc * 32;

    // ---- stage A tile (32 x 32) bf16, branch-free, conversion-free ----
    if constexpr (CIN % 4 == 0) {
      // all 4 k share (ky,kx); channel-consecutive -> one 8B load
      const int k  = k0 + a_kk;
      const int ky = k / (5 * CIN);
      const int r  = k - ky * (5 * CIN);
      const int kx = r / CIN;
      const int ci = r - kx * CIN;
      int iy = a_oy + ky - PAD;
      int ix = a_ox + kx - PAD;
      bool ok = true;
      if (PAD != 0) {
        ok = ((unsigned)iy < (unsigned)H) && ((unsigned)ix < (unsigned)W);
        iy = imin(imax(iy, 0), H - 1);
        ix = imin(imax(ix, 0), W - 1);
      }
      bf16x4 av = *(const bf16x4*)(a_base + ((long)iy * W + ix) * CIN + ci);
      const __bf16 zb = (__bf16)0.f;
      av[0] = ok ? av[0] : zb;
      av[1] = ok ? av[1] : zb;
      av[2] = ok ? av[2] : zb;
      av[3] = ok ? av[3] : zb;
      *(bf16x4*)&As[a_mi][a_kk] = av;
    } else {
#pragma unroll
      for (int i = 0; i < 4; ++i) {
        const int k   = k0 + a_kk + i;
        const int kcl = KG ? imin(k, KTOT - 1) : k;   // keep address in-range
        const int ky = kcl / (5 * CIN);
        const int r  = kcl - ky * (5 * CIN);
        const int kx = r / CIN;
        const int ci = r - kx * CIN;
        int iy = a_oy + ky - PAD;
        int ix = a_ox + kx - PAD;
        bool ok = (!KG || k < KTOT);
        if (PAD != 0) {
          ok = ok && ((unsigned)iy < (unsigned)H) && ((unsigned)ix < (unsigned)W);
          iy = imin(imax(iy, 0), H - 1);
          ix = imin(imax(ix, 0), W - 1);
        }
        const __bf16 t = a_base[((long)iy * W + ix) * CIN + ci];
        As[a_mi][a_kk + i] = ok ? t : (__bf16)0.f;
      }
    }
    __syncthreads();

    // ---- fragments ----
    const int mrow = wm * 16 + (lane & 15);
    const int kbA  = (lane < 16) ? 0 : 16;
    const bf16x16 af = cat8(*(const bf16x8*)&As[mrow][kbA],
                            *(const bf16x8*)&As[mrow][kbA + 8]);
    const bf16x16 bf0 = *(const bf16x16*)(bptr0 + k0);
    const bf16x16 bf1 = *(const bf16x16*)(bptr1 + k0);
    if (kc + 1 < KCH)   // gfx1250 global_prefetch_b8 of next weight chunk
      __builtin_prefetch(bptr0 + k0 + 32, 0, 1);

    acc0 = __builtin_amdgcn_wmma_f32_16x16x32_bf16(false, af, false, bf0,
                                                   (short)0, acc0, false, false);
    acc1 = __builtin_amdgcn_wmma_f32_16x16x32_bf16(false, af, false, bf1,
                                                   (short)0, acc1, false, false);
    __syncthreads();
  }

  // ---- epilogue: C layout -> global, optional bias / accumulate ----
  const int nloc  = wn * 32 + (lane & 15);
  const int mbase = m0 + wm * 16 + ((lane < 16) ? 0 : 8);
  const float bia0 = bias ? bias[n0 + nloc]      : 0.f;
  const float bia1 = bias ? bias[n0 + nloc + 16] : 0.f;
#pragma unroll
  for (int j = 0; j < 8; ++j) {
    const long o = (long)(mbase + j) * COUT + n0 + nloc;
    const float r0 = acc0[j] + bia0;
    const float r1 = acc1[j] + bia1;
    if (accum) {
      out[o]      += r0;
      out[o + 16] += r1;
    } else {
      out[o]      = r0;
      out[o + 16] = r1;
    }
  }
}

// ------------------------------ LSTM gates ---------------------------------
// z: (B*HoWo, 4F) fp32, gate order i,f,c,o.  c: fp32 state.  h written bf16
// with per-image stride (into a (B,T,...) sequence buffer).
__global__ void lstm_gate_kernel(const float* __restrict__ z,
                                 float* __restrict__ c,
                                 __bf16* __restrict__ hout, long h_img_stride,
                                 int HoWoF, int F, long n) {
  long idx = (long)blockIdx.x * blockDim.x + threadIdx.x;
  if (idx >= n) return;
  const long pg = idx / F;          // b*HoWo + p
  const int  f  = (int)(idx - pg * F);
  const long zb = pg * (long)(4 * F);
  const float zi = z[zb + f];
  const float zf = z[zb + F + f];
  const float zc = z[zb + 2 * F + f];
  const float zo = z[zb + 3 * F + f];
  const float ig = hard_sigmoid(zi);
  const float fg = hard_sigmoid(zf);
  const float og = hard_sigmoid(zo);
  const float cn = fg * c[idx] + ig * tanhf(zc);
  c[idx] = cn;
  const long b  = idx / HoWoF;
  const long wi = idx - b * HoWoF;
  hout[b * h_img_stride + wi] = (__bf16)(og * tanhf(cn));
}

// --------------------------- dense (M=16) GEMM -----------------------------
// part(split, 16, N) = A(16,K) @ W(K,N) over this split's K range.
// A is bf16; W is fp32 (streamed once, converted while staging).
// Block: 256 thr = 8 waves, each one 16-wide N sub-tile (block N tile 128).
// K must be a multiple of 32 (173056 and 1024 both are).
template <int N>
__global__ __launch_bounds__(256)
void dense_wmma(const __bf16* __restrict__ A,  // (16, K) bf16
                const float* __restrict__ Wt,  // (K, N) fp32
                float* __restrict__ part,      // (NSPLIT, 16, N)
                int K, int kchunks) {
  __shared__ __bf16 As[16][40];
  __shared__ __bf16 BsT[128][40];

  const int tid  = threadIdx.x;
  const int lane = tid & 31;
  const int wave = tid >> 5;
  const int n0   = blockIdx.x * 128;
  const int ks0  = blockIdx.y * kchunks;

  const int a_row = tid >> 4;        // 0..15
  const int a_col = (tid & 15) * 2;  // 0,2,...,30
  const int b_n   = tid & 127;
  const int b_kh  = (tid >> 7) * 16;

  f32x8 acc = {};

  for (int kc = 0; kc < kchunks; ++kc) {
    const int k0 = (ks0 + kc) * 32;
    // stage A (16 x 32): two consecutive bf16 per thread, pure copy
    {
      *(bf16x2*)&As[a_row][a_col] =
          *(const bf16x2*)(A + (long)a_row * K + k0 + a_col);
    }
    // stage B (32 x 128) -> BsT (n, k)
    {
      const float* wcol = Wt + (long)(k0 + b_kh) * N + (n0 + b_n);
      if (kc + 1 < kchunks)
        __builtin_prefetch(wcol + (long)32 * N, 0, 1);
      bf16x16 bv;
#pragma unroll
      for (int i = 0; i < 16; ++i) bv[i] = (__bf16)wcol[(long)i * N];
      *(bf16x16*)&BsT[b_n][b_kh] = bv;
    }
    __syncthreads();

    const int mrow = lane & 15;
    const int kbA  = (lane < 16) ? 0 : 16;
    const bf16x16 af = cat8(*(const bf16x8*)&As[mrow][kbA],
                            *(const bf16x8*)&As[mrow][kbA + 8]);
    const int ncol = wave * 16 + (lane & 15);
    const int kbB  = (lane < 16) ? 0 : 16;
    const bf16x16 bfr = *(const bf16x16*)&BsT[ncol][kbB];

    acc = __builtin_amdgcn_wmma_f32_16x16x32_bf16(false, af, false, bfr,
                                                  (short)0, acc, false, false);
    __syncthreads();
  }

  const int n     = n0 + wave * 16 + (lane & 15);
  const int mbase = (lane < 16) ? 0 : 8;
#pragma unroll
  for (int j = 0; j < 8; ++j)
    part[((long)blockIdx.y * 16 + mbase + j) * N + n] = acc[j];
}

__global__ void reduce_bias_relu(const float* __restrict__ part,
                                 const float* __restrict__ bias,
                                 __bf16* __restrict__ out,
                                 int N, int nsplit, int do_relu) {
  const int idx = blockIdx.x * blockDim.x + threadIdx.x;
  if (idx >= 16 * N) return;
  const int m = idx / N, n = idx - m * N;
  float s = bias[n];
  for (int sp = 0; sp < nsplit; ++sp) s += part[((long)sp * 16 + m) * N + n];
  if (do_relu) s = fmaxf(s, 0.f);
  out[idx] = (__bf16)s;
}

__global__ void dense3_kernel(const __bf16* __restrict__ A,  // (16,1024) bf16
                              const float* __restrict__ W,   // (1024,4)
                              const float* __restrict__ b,   // (4)
                              float* __restrict__ out) {     // (16,4)
  const int o = threadIdx.x;  // 0..63
  const int m = o >> 2, n = o & 3;
  float s = b[n];
  for (int k = 0; k < 1024; ++k) s += (float)A[m * 1024 + k] * W[k * 4 + n];
  out[o] = s;
}

// ------------------------------- launcher ----------------------------------
extern "C" void kernel_launch(void* const* d_in, const int* in_sizes, int n_in,
                              void* d_out, int out_size, void* d_ws, size_t ws_size,
                              hipStream_t stream) {
  (void)in_sizes; (void)n_in; (void)out_size; (void)ws_size;
  const float* x   = (const float*)d_in[0];
  const float* Wx1 = (const float*)d_in[1];
  const float* Wh1 = (const float*)d_in[2];
  const float* b1  = (const float*)d_in[3];
  const float* Wx2 = (const float*)d_in[4];
  const float* Wh2 = (const float*)d_in[5];
  const float* b2  = (const float*)d_in[6];
  const float* Wx3 = (const float*)d_in[7];
  const float* Wh3 = (const float*)d_in[8];
  const float* b3  = (const float*)d_in[9];
  const float* Wd1 = (const float*)d_in[10];
  const float* bd1 = (const float*)d_in[11];
  const float* Wd2 = (const float*)d_in[12];
  const float* bd2 = (const float*)d_in[13];
  const float* Wd3 = (const float*)d_in[14];
  const float* bd3 = (const float*)d_in[15];
  float* outp = (float*)d_out;

  const int B = 16, T = 6;
  const int HW1 = 3600, F1 = 128;   // 60x60
  const int HW2 = 3136, F2 = 64;    // 56x56
  const int HW3 = 2704, F3 = 64;    // 52x52
  const long NX = (long)B * T * 64 * 64 * 5;   // 1,966,080

  // ---- workspace layout (alloc unit = float, all sizes multiples of 4) ----
  float* w = (float*)d_ws;
  size_t off = 0;
  auto alloc = [&](size_t nf) { float* p = w + off; off += nf; return p; };
  float*  zero_h = alloc((size_t)B * HW1 * F1);          // stays zero (fp32/bf16)
  float*  c1     = alloc((size_t)B * HW1 * F1);
  float*  c2     = alloc((size_t)B * HW2 * F2);
  float*  c3     = alloc((size_t)B * HW3 * F3);
  float*  zbuf   = alloc((size_t)B * HW1 * 4 * F1);      // max z (fp32)
  float*  dpart  = alloc((size_t)32 * 16 * 1024);
  __bf16* xbf    = (__bf16*)alloc((size_t)NX / 2);
  __bf16* hseq1  = (__bf16*)alloc((size_t)B * T * HW1 * F1 / 2);
  __bf16* hseq2  = (__bf16*)alloc((size_t)B * T * HW2 * F2 / 2);
  __bf16* h3a    = (__bf16*)alloc((size_t)B * HW3 * F3 / 2);
  __bf16* h3b    = (__bf16*)alloc((size_t)B * HW3 * F3 / 2);
  __bf16* a1     = (__bf16*)alloc(16 * 1024 / 2);
  __bf16* a2     = (__bf16*)alloc(16 * 1024 / 2);
  // packed bf16 conv weights (COUT, K32)
  __bf16* wp_x1 = (__bf16*)alloc((size_t)512 * 128  / 2);
  __bf16* wp_h1 = (__bf16*)alloc((size_t)512 * 3200 / 2);
  __bf16* wp_x2 = (__bf16*)alloc((size_t)256 * 3200 / 2);
  __bf16* wp_h2 = (__bf16*)alloc((size_t)256 * 1600 / 2);
  __bf16* wp_x3 = (__bf16*)alloc((size_t)256 * 1600 / 2);
  __bf16* wp_h3 = (__bf16*)alloc((size_t)256 * 1600 / 2);
  __bf16* h3[2] = {h3a, h3b};
  const __bf16* zh = (const __bf16*)zero_h;

  // zero-init zero_h + c1 + c2 + c3 (contiguous region)
  {
    long nz = (long)(B * HW1 * F1) * 2 + (long)B * HW2 * F2 + (long)B * HW3 * F3;
    fill_zero_kernel<<<dim3((unsigned)((nz + 255) / 256)), 256, 0, stream>>>(zero_h, nz);
  }
  // x -> bf16
  f32_to_bf16_kernel<<<dim3((unsigned)((NX + 255) / 256)), 256, 0, stream>>>(
      x, xbf, NX);
  // pack conv weights -> bf16 (COUT, K32)
  auto pack = [&](const float* src, __bf16* dst, int KT, int K32, int CO) {
    long ne = (long)CO * K32;
    pack_weight_kernel<<<dim3((unsigned)((ne + 255) / 256)), 256, 0, stream>>>(
        src, dst, KT, K32, CO, ne);
  };
  pack(Wx1, wp_x1, 125, 128, 512);
  pack(Wh1, wp_h1, 3200, 3200, 512);
  pack(Wx2, wp_x2, 3200, 3200, 256);
  pack(Wh2, wp_h2, 1600, 1600, 256);
  pack(Wx3, wp_x3, 1600, 1600, 256);
  pack(Wh3, wp_h3, 1600, 1600, 256);

  // ------------------------------ layer 1 ------------------------------
  for (int t = 0; t < T; ++t) {
    dim3 g((unsigned)(B * HW1 / 32), 512 / 128);
    const __bf16* xt = xbf + (size_t)t * 64 * 64 * 5;
    conv5x5_wmma<5, 512, 0><<<g, 256, 0, stream>>>(
        xt, (long)T * 64 * 64 * 5, wp_x1, b1, zbuf, 64, 64, 60, 60, 0);
    const __bf16* hp = t ? hseq1 + (size_t)(t - 1) * HW1 * F1 : zh;
    const long hstr = t ? (long)T * HW1 * F1 : (long)HW1 * F1;
    conv5x5_wmma<128, 512, 2><<<g, 256, 0, stream>>>(
        hp, hstr, wp_h1, nullptr, zbuf, 60, 60, 60, 60, 1);
    long n = (long)B * HW1 * F1;
    lstm_gate_kernel<<<dim3((unsigned)((n + 255) / 256)), 256, 0, stream>>>(
        zbuf, c1, hseq1 + (size_t)t * HW1 * F1, (long)T * HW1 * F1,
        HW1 * F1, F1, n);
  }

  // ------------------------------ layer 2 ------------------------------
  for (int t = 0; t < T; ++t) {
    dim3 g((unsigned)(B * HW2 / 32), 256 / 128);
    conv5x5_wmma<128, 256, 0><<<g, 256, 0, stream>>>(
        hseq1 + (size_t)t * HW1 * F1, (long)T * HW1 * F1, wp_x2, b2, zbuf,
        60, 60, 56, 56, 0);
    const __bf16* hp = t ? hseq2 + (size_t)(t - 1) * HW2 * F2 : zh;
    const long hstr = t ? (long)T * HW2 * F2 : (long)HW2 * F2;
    conv5x5_wmma<64, 256, 2><<<g, 256, 0, stream>>>(
        hp, hstr, wp_h2, nullptr, zbuf, 56, 56, 56, 56, 1);
    long n = (long)B * HW2 * F2;
    lstm_gate_kernel<<<dim3((unsigned)((n + 255) / 256)), 256, 0, stream>>>(
        zbuf, c2, hseq2 + (size_t)t * HW2 * F2, (long)T * HW2 * F2,
        HW2 * F2, F2, n);
  }

  // ------------------------------ layer 3 ------------------------------
  for (int t = 0; t < T; ++t) {
    dim3 g((unsigned)(B * HW3 / 32), 256 / 128);
    conv5x5_wmma<64, 256, 0><<<g, 256, 0, stream>>>(
        hseq2 + (size_t)t * HW2 * F2, (long)T * HW2 * F2, wp_x3, b3, zbuf,
        56, 56, 52, 52, 0);
    const __bf16* hp = t ? h3[(t - 1) & 1] : zh;
    conv5x5_wmma<64, 256, 2><<<g, 256, 0, stream>>>(
        hp, (long)HW3 * F3, wp_h3, nullptr, zbuf, 52, 52, 52, 52, 1);
    long n = (long)B * HW3 * F3;
    lstm_gate_kernel<<<dim3((unsigned)((n + 255) / 256)), 256, 0, stream>>>(
        zbuf, c3, h3[t & 1], (long)HW3 * F3, HW3 * F3, F3, n);
  }
  __bf16* hfin = h3[(T - 1) & 1];  // (16, 52*52*64) = (16, 173056) row-major

  // ------------------------------- head --------------------------------
  // dense1: K=173056 = 32 splits * 169 chunks * 32
  dense_wmma<1024><<<dim3(1024 / 128, 32), 256, 0, stream>>>(hfin, Wd1, dpart,
                                                             173056, 169);
  reduce_bias_relu<<<dim3(16 * 1024 / 256), 256, 0, stream>>>(dpart, bd1, a1,
                                                              1024, 32, 1);
  // dense2: K=1024 = 1 split * 32 chunks
  dense_wmma<1024><<<dim3(1024 / 128, 1), 256, 0, stream>>>(a1, Wd2, dpart,
                                                            1024, 32);
  reduce_bias_relu<<<dim3(16 * 1024 / 256), 256, 0, stream>>>(dpart, bd2, a2,
                                                              1024, 1, 1);
  // dense3: (16,1024)@(1024,4)+b -> d_out
  dense3_kernel<<<1, 64, 0, stream>>>(a2, Wd3, bd3, outp);
}
